// PredRNN_Plus_22539988369573
// MI455X (gfx1250) — matrix operand order repaired and checked by reading, not compiled, not run on hardware
//
#include <hip/hip_runtime.h>
#include <hip/hip_bf16.h>
#include <math.h>

// ---------------------------------------------------------------------------
// PredRNN++ forward for MI455X (gfx1250, wave32).
// All convs -> implicit GEMM on v_wmma_f32_16x16x32_f16 (f16 in, f32 acc).
// Weights pre-reordered to k' = tap*Cin + c (f16, zero-padded to Kpad).
// Every conv input is kept as an f16 NHWC image (written by its producer), so
// the GEMM B-stage (im2col) is two contiguous 16B vector loads per thread.
// ---------------------------------------------------------------------------

typedef __attribute__((ext_vector_type(16))) _Float16 v16h;
typedef __attribute__((ext_vector_type(8)))  _Float16 h8;
typedef __attribute__((ext_vector_type(8)))  float    v8f;

#define NB    4
#define HW    1024          // 32*32
#define NH_   64
#define TSTEPS 9

// ---------------- utility kernels ----------------

__global__ void k_zero(float* __restrict__ p, int n) {
  int i = blockIdx.x * 256 + threadIdx.x;
  if (i < n) p[i] = 0.f;
}

__global__ void k_zeroh(_Float16* __restrict__ p, int n) {
  int i = blockIdx.x * 256 + threadIdx.x;
  if (i < n) p[i] = (_Float16)0.f;
}

// Reorder OIHW f32 weights -> f16 [Cout][Kpad], k' = tap*Cin + c, zero padded.
__global__ void k_wprep(const float* __restrict__ src, _Float16* __restrict__ dst,
                        int cinLog2, int taps, int K, int Kpad) {
  int k = blockIdx.x * 256 + threadIdx.x;
  int o = blockIdx.y;
  if (k >= Kpad) return;
  _Float16 v = (_Float16)0.f;
  if (k < K) {
    int tap = k >> cinLog2;
    int c   = k & ((1 << cinLog2) - 1);
    v = (_Float16)src[(((size_t)o << cinLog2) + c) * taps + tap];
  }
  dst[(size_t)o * Kpad + k] = v;
}

// ---------------- implicit-GEMM conv via WMMA ----------------
// Y[Cout, N=4096] = W[Cout, Kpad] x im2col(Xh)[Kpad, N]  (+bias)
// Xh: f16 NHWC [b][y][x][c].  Block: 256 threads = 8 wave32 as 2(M)x4(N);
// tile 64(M) x 128(N); each wave: 2 A-frags x 2 B-frags -> 4 v_wmma per chunk.
__global__ __launch_bounds__(256)
void k_conv_gemm(const _Float16* __restrict__ Xh, const _Float16* __restrict__ W,
                 const float* __restrict__ bias, float* __restrict__ Y,
                 int cinLog2, int Cout, int K, int Kpad, int five) {
  __shared__ __attribute__((aligned(32))) _Float16 lA[64 * 32];   // [m][k]
  __shared__ __attribute__((aligned(32))) _Float16 lB[128 * 32];  // [n][k]

  const int tid  = threadIdx.x;
  const int lane = tid & 31;
  const int wave = tid >> 5;
  const int wvm  = wave & 1;   // 2 waves along M (32 each)
  const int wvn  = wave >> 1;  // 4 waves along N (32 each)
  const int m0   = blockIdx.x * 64;
  const int n0   = blockIdx.y * 128;

  v8f acc[2][2];
#pragma unroll
  for (int i = 0; i < 2; ++i)
#pragma unroll
    for (int j = 0; j < 2; ++j)
#pragma unroll
      for (int r = 0; r < 8; ++r) acc[i][j][r] = 0.f;

  // A-load mapping: 64 rows x 32 k, 8 halfs per thread
  const int ar = tid >> 2;          // 0..63
  const int aq = (tid & 3) << 3;    // 0,8,16,24
  const int gm = m0 + ar;

  // B-load mapping: 128 pixels x 32 k, one 16-k run per thread
  const int bp  = tid >> 1;         // 0..127
  const int bk  = (tid & 1) << 4;   // 0 or 16
  const int gn  = n0 + bp;
  const int bb  = gn >> 10;
  const int pix = gn & (HW - 1);
  const int py  = pix >> 5, px = pix & 31;

  for (int kc = 0; kc < Kpad; kc += 32) {
    // ---- stage A (padded f16 weights; no K guard needed) ----
    h8 av;
    if (gm < Cout) {
      const _Float16* wp = W + (size_t)gm * Kpad + kc + aq;
      av = *reinterpret_cast<const h8*>(wp);
      __builtin_prefetch(wp + 32, 0, 3);  // global_prefetch_b8, next chunk
    } else {
#pragma unroll
      for (int i = 0; i < 8; ++i) av[i] = (_Float16)0.f;
    }
    *reinterpret_cast<h8*>(&lA[ar * 32 + aq]) = av;

    // ---- stage B: one im2col run of 16 consecutive k' (single tap) ----
    // k' = tap*Cin + c; run is 16-aligned and Cin is a power of two >= 16,
    // so tap/iy/ix are constant over the run. NHWC: the 16 channels are
    // contiguous halfs -> two 16B vector loads, no per-element work.
    h8 blo, bhi;
    {
      const int kb = kc + bk;
      int iy = py, ix = px, c0 = kb;
      if (five) {
        int tap = kb >> cinLog2;
        c0 = kb & ((1 << cinLog2) - 1);
        int fy = tap / 5;
        int fx = tap - fy * 5;
        iy = py + fy - 2;
        ix = px + fx - 2;
      }
      if (kb < K && (unsigned)iy < 32u && (unsigned)ix < 32u) {
        const _Float16* xp =
            Xh + (((size_t)bb * HW + (iy << 5) + ix) << cinLog2) + c0;
        blo = *reinterpret_cast<const h8*>(xp);
        bhi = *reinterpret_cast<const h8*>(xp + 8);
      } else {
#pragma unroll
        for (int t = 0; t < 8; ++t) { blo[t] = (_Float16)0.f; bhi[t] = (_Float16)0.f; }
      }
    }
    *reinterpret_cast<h8*>(&lB[bp * 32 + bk]) = blo;
    *reinterpret_cast<h8*>(&lB[bp * 32 + bk + 8]) = bhi;
    __syncthreads();

    // ---- fragments: A and B both use contiguous k = L16*16 + e ----
    const int L16 = lane >> 4;
    v16h af[2], bf[2];
#pragma unroll
    for (int i = 0; i < 2; ++i) {
      // A: e=0..15 -> k = L16*16 + e
      int am = wvm * 32 + i * 16 + (lane & 15);
      h8 alo = *reinterpret_cast<const h8*>(&lA[am * 32 + L16 * 16]);
      h8 ahi = *reinterpret_cast<const h8*>(&lA[am * 32 + L16 * 16 + 8]);
      af[i] = __builtin_shufflevector(alo, ahi, 0, 1, 2, 3, 4, 5, 6, 7,
                                      8, 9, 10, 11, 12, 13, 14, 15);
    }
#pragma unroll
    for (int j = 0; j < 2; ++j) {
      // B: n = lane&15, e=0..15 -> k = L16*16 + e (contiguous in LDS)
      int bn = wvn * 32 + j * 16 + (lane & 15);
      bf[j] = *reinterpret_cast<const v16h*>(&lB[bn * 32 + L16 * 16]);
    }
#pragma unroll
    for (int i = 0; i < 2; ++i)
#pragma unroll
      for (int j = 0; j < 2; ++j)
        acc[i][j] = __builtin_amdgcn_wmma_f32_16x16x32_f16(
            false, af[i], false, bf[j], (short)0, acc[i][j], false, false);
    __syncthreads();
  }

  // ---- store C (m = r + (lane/16)*8, n = lane&15) + bias ----
#pragma unroll
  for (int i = 0; i < 2; ++i)
#pragma unroll
    for (int j = 0; j < 2; ++j) {
      int n = n0 + wvn * 32 + j * 16 + (lane & 15);
      int b = n >> 10;
      int p = n & (HW - 1);
#pragma unroll
      for (int r = 0; r < 8; ++r) {
        int m = m0 + wvm * 32 + i * 16 + (lane >> 4) * 8 + r;
        if (m < Cout) {
          float o = acc[i][j][r] + (bias ? bias[m] : 0.f);
          Y[(((size_t)b * Cout + m) << 10) + p] = o;
        }
      }
    }
}

// ---------------- elementwise kernels ----------------
// Conv outputs stay f32 NCHW; recurrent conv inputs are written as f16 NHWC.

__device__ __forceinline__ float sigm(float x) { return 1.f / (1.f + __expf(-x)); }

// i = sig(ix+ih+ic); f = sig(fx+fh+fc+1); g = tanh(gx+gh+gc); c = f*c + i*g
__global__ void k_gates1(const float* __restrict__ xc, const float* __restrict__ hc,
                         const float* __restrict__ cc, float* __restrict__ c,
                         _Float16* __restrict__ chh, _Float16* __restrict__ cmh) {
  int i = blockIdx.x * 256 + threadIdx.x;
  if (i >= NB * NH_ * HW) return;
  int b = i >> 16, cp = i & 65535, ch = cp >> 10, p = cp & 1023;
  size_t xb = ((size_t)b * 448 + ch) * HW + p;
  size_t hb = ((size_t)b * 256 + ch) * HW + p;
  size_t cb = ((size_t)b * 192 + ch) * HW + p;
  float ig = sigm(xc[xb] + hc[hb] + cc[cb]);
  float fg = sigm(xc[xb + 128 * HW] + hc[hb + 128 * HW] + cc[cb + 128 * HW] + 1.f);
  float g  = tanhf(xc[xb + 64 * HW] + hc[hb + 64 * HW] + cc[cb + 64 * HW]);
  size_t ci = ((size_t)b * NH_ + ch) * HW + p;
  float cn = fg * c[ci] + ig * g;
  c[ci] = cn;
  size_t nh = (size_t)b * HW + p;               // NHWC pixel index
  chh[(nh << 6) + ch] = (_Float16)cn;           // 64-ch image
  cmh[(nh << 7) + ch] = (_Float16)cn;           // cellmem[:, :64]
}

// ii = sig(icp+ixp+im); ff = sig(fcp+fxp+fm+1); gg = tanh(gcp+gxp)
// m = ff*tanh(mm) + ii*gg
__global__ void k_gates2(const float* __restrict__ c2m, const float* __restrict__ xc,
                         const float* __restrict__ mc, _Float16* __restrict__ memh,
                         _Float16* __restrict__ cmh) {
  int i = blockIdx.x * 256 + threadIdx.x;
  if (i >= NB * NH_ * HW) return;
  int b = i >> 16, cp = i & 65535, ch = cp >> 10, p = cp & 1023;
  size_t c2 = ((size_t)b * 256 + ch) * HW + p;
  size_t xb = ((size_t)b * 448 + ch) * HW + p;
  size_t mb = ((size_t)b * 192 + ch) * HW + p;
  float ii = sigm(c2m[c2] + xc[xb + 256 * HW] + mc[mb]);
  float ff = sigm(c2m[c2 + 128 * HW] + xc[xb + 384 * HW] + mc[mb + 64 * HW] + 1.f);
  float gg = tanhf(c2m[c2 + 64 * HW] + xc[xb + 320 * HW]);
  float mn = ff * tanhf(mc[mb + 128 * HW]) + ii * gg;
  size_t nh = (size_t)b * HW + p;
  memh[(nh << 6) + ch] = (_Float16)mn;
  cmh[(nh << 7) + 64 + ch] = (_Float16)mn;      // cellmem[:, 64:]
}

// o = tanh(ox+oh+oc+om); h = o * tanh(lastout)
__global__ void k_gates3(const float* __restrict__ xc, const float* __restrict__ hc,
                         const float* __restrict__ c2m, const float* __restrict__ om,
                         const float* __restrict__ lastout, _Float16* __restrict__ hh) {
  int i = blockIdx.x * 256 + threadIdx.x;
  if (i >= NB * NH_ * HW) return;
  int b = i >> 16, cp = i & 65535, ch = cp >> 10, p = cp & 1023;
  size_t xb = ((size_t)b * 448 + 192 + ch) * HW + p;
  size_t hb = ((size_t)b * 256 + 192 + ch) * HW + p;
  size_t c2 = ((size_t)b * 256 + 192 + ch) * HW + p;
  size_t ci = ((size_t)b * NH_ + ch) * HW + p;
  float o = tanhf(xc[xb] + hc[hb] + c2m[c2] + om[ci]);
  hh[(((size_t)b * HW + p) << 6) + ch] = (_Float16)(o * tanhf(lastout[ci]));
}

// pg = tanh(gx0+gz0); ug = sig(gx1+gz1); z = ug*pg + (1-ug)*z
__global__ void k_ghu(const float* __restrict__ gx, const float* __restrict__ gz,
                      float* __restrict__ z, _Float16* __restrict__ zh) {
  int i = blockIdx.x * 256 + threadIdx.x;
  if (i >= NB * NH_ * HW) return;
  int b = i >> 16, cp = i & 65535, ch = cp >> 10, p = cp & 1023;
  size_t gb = ((size_t)b * 128 + ch) * HW + p;
  float pg = tanhf(gx[gb] + gz[gb]);
  float ug = sigm(gx[gb + 64 * HW] + gz[gb + 64 * HW]);
  size_t zi = ((size_t)b * NH_ + ch) * HW + p;
  float zn = ug * pg + (1.f - ug) * z[zi];
  z[zi] = zn;
  zh[(((size_t)b * HW + p) << 6) + ch] = (_Float16)zn;
}

// net = (t<5) ? frame : mask*frame + (1-mask)*x_gen  -> f16 NHWC only
__global__ void k_make_input(const float* __restrict__ frames, const float* __restrict__ mask,
                             const float* __restrict__ xgen, _Float16* __restrict__ neth,
                             int t) {
  int i = blockIdx.x * 256 + threadIdx.x;
  if (i >= NB * 16 * HW) return;
  int b = i >> 14, cp = i & 16383, ch = cp >> 10, p = cp & 1023;
  float fr = frames[(((size_t)b * 10 + t) * HW + p) * 16 + ch];
  float v;
  if (t < 5) {
    v = fr;
  } else {
    float m  = mask[(((size_t)b * 4 + (t - 5)) * HW + p) * 16 + ch];
    float xg = xgen[((size_t)b * 16 + ch) * HW + p];
    v = m * fr + (1.f - m) * xg;
  }
  neth[(((size_t)b * HW + p) << 4) + ch] = (_Float16)v;
}

// d_out[b, t, y, x, c] = x_gen[b, c, y, x]
__global__ void k_out_store(const float* __restrict__ xgen, float* __restrict__ out, int t) {
  int i = blockIdx.x * 256 + threadIdx.x;
  if (i >= NB * 16 * HW) return;
  int b = i >> 14, cp = i & 16383, ch = cp >> 10, p = cp & 1023;
  out[(((size_t)b * TSTEPS + t) * HW + p) * 16 + ch] = xgen[((size_t)b * 16 + ch) * HW + p];
}

// ---------------- host side ----------------

enum { WX, BX, WH, BH, WC, BC, WM, BM, WC2M, BC2M, WOM, BOM, WLAST, BLAST };

extern "C" void kernel_launch(void* const* d_in, const int* in_sizes, int n_in,
                              void* d_out, int out_size, void* d_ws, size_t ws_size,
                              hipStream_t stream) {
  (void)n_in; (void)out_size; (void)ws_size;
  const float* frames = (const float*)d_in[0];
  const float* maskt  = (const float*)d_in[1];

  // Detect pytree flatten order: insertion-order (cells[0].wx first, 179200)
  // vs JAX sorted-key order (cells[0].bc first, 192).
  const bool sorted = (in_sizes[2] == 192);
  static const int insM[14] = {0, 1, 2, 3, 4, 5, 6, 7, 8, 9, 10, 11, 12, 13};
  static const int srtM[14] = {13, 6, 9, 2, 7, 0, 11, 4, 8, 1, 12, 5, 10, 3};
  static const int insG[4]  = {0, 1, 2, 3};   // wz, bz, wx, bx
  static const int srtG[4]  = {3, 1, 2, 0};
  const int* M = sorted ? srtM : insM;
  const int* G = sorted ? srtG : insG;

  const float* cw[4][14];
  int idx = 2;
  for (int l = 0; l < 4; ++l) {
    for (int s = 0; s < 14; ++s) cw[l][s] = (const float*)d_in[idx + M[s]];
    idx += 14;
  }
  const float* ghu_wz = (const float*)d_in[idx + G[0]];
  const float* ghu_bz = (const float*)d_in[idx + G[1]];
  const float* ghu_wx = (const float*)d_in[idx + G[2]];
  const float* ghu_bx = (const float*)d_in[idx + G[3]];
  idx += 4;
  const float* wout = (const float*)d_in[idx];

  // ---- workspace bump allocator (256B aligned) ----
  char* ws = (char*)d_ws;
  size_t off = 0;
  auto alloc = [&](size_t bytes) -> void* {
    off = (off + 255) & ~(size_t)255;
    void* p = ws + off;
    off += bytes;
    return p;
  };
  auto allocF = [&](size_t elems) -> float* { return (float*)alloc(elems * 4); };
  auto allocH = [&](size_t elems) -> _Float16* { return (_Float16*)alloc(elems * 2); };

  const size_t SP = (size_t)NB * HW;  // 4096 pixels

  // f32 buffers (elementwise recurrences + conv outputs, NCHW)
  float* cbuf[4];
  for (int l = 0; l < 4; ++l) cbuf[l] = allocF(SP * NH_);
  float* zbuf    = allocF(SP * NH_);
  float* xgen    = allocF(SP * 16);
  float* xc      = allocF(SP * 448);
  float* hc      = allocF(SP * 256);
  float* cc      = allocF(SP * 192);
  float* mc      = allocF(SP * 192);
  float* c2m     = allocF(SP * 256);
  float* om      = allocF(SP * NH_);
  float* lastout = allocF(SP * NH_);
  float* gx      = allocF(SP * 128);
  float* gz      = allocF(SP * 128);

  // f16 NHWC conv-input images
  _Float16* hh[4]; _Float16* chh[4];
  for (int l = 0; l < 4; ++l) hh[l]  = allocH(SP * NH_);
  for (int l = 0; l < 4; ++l) chh[l] = allocH(SP * NH_);
  _Float16* memh = allocH(SP * NH_);
  _Float16* zh   = allocH(SP * NH_);
  _Float16* neth = allocH(SP * 16);
  _Float16* cmh  = allocH(SP * 128);  // cellmem = concat(c_new, m_new)

  auto log2i = [](int c) { return c == 16 ? 4 : (c == 64 ? 6 : 7); };

  // weight prep: reorder (o,c,tap)->(o, tap*Cin+c), f16, zero-pad K to 32
  auto prep = [&](const float* s, int Cout_, int Cin_, int taps) -> _Float16* {
    int K = Cin_ * taps, Kpad = (K + 31) & ~31;
    _Float16* d = allocH((size_t)Cout_ * Kpad);
    dim3 g((Kpad + 255) / 256, Cout_);
    k_wprep<<<g, 256, 0, stream>>>(s, d, log2i(Cin_), taps, K, Kpad);
    return d;
  };

  _Float16 *wxf[4], *whf[4], *wcf[4], *wmf[4], *wc2mf[4], *womf[4], *wlastf[4];
  for (int l = 0; l < 4; ++l) {
    int cin = (l == 0) ? 16 : 64;
    wxf[l]    = prep(cw[l][WX],    448, cin, 25);
    whf[l]    = prep(cw[l][WH],    256, 64,  25);
    wcf[l]    = prep(cw[l][WC],    192, 64,  25);
    wmf[l]    = prep(cw[l][WM],    192, 64,  25);
    wc2mf[l]  = prep(cw[l][WC2M],  256, 64,  25);
    womf[l]   = prep(cw[l][WOM],    64, 64,  25);
    wlastf[l] = prep(cw[l][WLAST],  64, 128, 1);
  }
  _Float16* wzf   = prep(ghu_wz, 128, 64, 25);
  _Float16* wxgf  = prep(ghu_wx, 128, 64, 25);
  _Float16* woutf = prep(wout,    16, 64, 1);

  // zero recurrent state (both precisions)
  auto zeroF = [&](float* p, int n) { k_zero<<<(n + 255) / 256, 256, 0, stream>>>(p, n); };
  auto zeroH = [&](_Float16* p, int n) { k_zeroh<<<(n + 255) / 256, 256, 0, stream>>>(p, n); };
  for (int l = 0; l < 4; ++l) {
    zeroF(cbuf[l], SP * NH_);
    zeroH(hh[l], SP * NH_);
    zeroH(chh[l], SP * NH_);
  }
  zeroF(zbuf, SP * NH_);
  zeroF(xgen, SP * 16);
  zeroH(memh, SP * NH_);
  zeroH(zh, SP * NH_);

  auto gemm = [&](const _Float16* Xh, const _Float16* W, const float* bias, float* Y,
                  int Cin, int Cout, bool five) {
    int K = five ? Cin * 25 : Cin;
    int Kpad = (K + 31) & ~31;
    dim3 g((Cout + 63) / 64, 32);  // N = 4096 -> 32 tiles of 128
    k_conv_gemm<<<g, 256, 0, stream>>>(Xh, W, bias, Y, log2i(Cin), Cout, K, Kpad,
                                       five ? 1 : 0);
  };

  const int eltB = (NB * NH_ * HW) / 256;  // 1024
  const int ioB  = (NB * 16 * HW) / 256;   // 256

  for (int t = 0; t < TSTEPS; ++t) {
    k_make_input<<<ioB, 256, 0, stream>>>(frames, maskt, xgen, neth, t);

    for (int l = 0; l < 4; ++l) {
      const _Float16* xin; int cin;
      if (l == 0)      { xin = neth;       cin = 16; }
      else if (l == 1) { xin = zh;         cin = 64; }
      else             { xin = hh[l - 1];  cin = 64; }

      gemm(xin,     wxf[l], cw[l][BX], xc, cin, 448, true);
      gemm(hh[l],   whf[l], cw[l][BH], hc, 64,  256, true);
      gemm(chh[l],  wcf[l], cw[l][BC], cc, 64,  192, true);
      gemm(memh,    wmf[l], cw[l][BM], mc, 64,  192, true);
      k_gates1<<<eltB, 256, 0, stream>>>(xc, hc, cc, cbuf[l], chh[l], cmh);
      gemm(chh[l], wc2mf[l], cw[l][BC2M], c2m, 64, 256, true);
      k_gates2<<<eltB, 256, 0, stream>>>(c2m, xc, mc, memh, cmh);
      gemm(memh, womf[l],   cw[l][BOM],   om,      64,  64, true);
      gemm(cmh,  wlastf[l], cw[l][BLAST], lastout, 128, 64, false);
      k_gates3<<<eltB, 256, 0, stream>>>(xc, hc, c2m, om, lastout, hh[l]);

      if (l == 0) {  // GHU between layer 0 and layer 1
        gemm(hh[0], wxgf, ghu_bx, gx, 64, 128, true);
        gemm(zh,    wzf,  ghu_bz, gz, 64, 128, true);
        k_ghu<<<eltB, 256, 0, stream>>>(gx, gz, zbuf, zh);
      }
    }

    gemm(hh[3], woutf, nullptr, xgen, 64, 16, false);  // 1x1, no bias
    k_out_store<<<ioB, 256, 0, stream>>>(xgen, (float*)d_out, t);
  }
}
